// PartialEncoderEDDIATSE_57767310131606
// MI455X (gfx1250) — matrix operand order repaired and checked by reading, not compiled, not run on hardware
//
#include <hip/hip_runtime.h>
#include <hip/hip_bf16.h>

// ---------------------------------------------------------------------------
// Problem constants (from reference setup_inputs)
// ---------------------------------------------------------------------------
#define B_   32
#define J_   16384
#define D_   32
#define AE_  16
#define H_   128
#define EH_  256
#define LAT_ 32
#define INDIM_ 49          // 1 + D + Ae
#define EPS_ 1e-5f

#define TILE_ROWS 128      // rows (b,j pairs) per block
#define THREADS   256      // 8 wave32 per block

// LDS strides (bf16 elements), padded for bank-conflict avoidance, 16B aligned
#define STRIDE_A  72       // K=64 + 8 pad   (144 bytes/row)
#define STRIDE_H  136      // K=128 + 8 pad  (272 bytes/row)

// LDS byte offsets (A/W1 region aliased with H1 region after a barrier)
#define OFF_SA   0                         // 128*72*2  = 18432
#define OFF_SW1  18432                     // 128*72*2  = 18432 -> 36864
#define OFF_SH1  0                         // 128*136*2 = 34816 (alias of SA+SW1)
#define OFF_SW2  36864                     // 32*136*2  = 8704  -> 45568
#define OFF_F32  45568
#define SMEM_BYTES 49152

typedef __attribute__((ext_vector_type(16))) __bf16 v16bf;
typedef __attribute__((ext_vector_type(8)))  float  v8f;

union Frag { v16bf v; uint4 u[2]; };

// p points at first element of the low 8-element chunk; hiOff = element offset
// of the high chunk (A-frag and B-frag: 16).
__device__ __forceinline__ v16bf load_frag(const __bf16* p, int hiOff) {
    Frag f;
    f.u[0] = *(const uint4*)(p);
    f.u[1] = *(const uint4*)(p + hiOff);
    return f.v;
}

__device__ __forceinline__ v8f wmma_bf16(v16bf a, v16bf b, v8f c) {
    // D = A(16x32) * B(32x16) + C, fp32 accumulate
    return __builtin_amdgcn_wmma_f32_16x16x32_bf16(false, a, false, b,
                                                   (short)0, c, false, false);
}

// ---------------------------------------------------------------------------
// Kernel 0: zero the pooled accumulator + counts in workspace
// ---------------------------------------------------------------------------
__global__ void zero_ws_kernel(float* ws, int n) {
    int i = blockIdx.x * blockDim.x + threadIdx.x;
    if (i < n) ws[i] = 0.0f;
}

// ---------------------------------------------------------------------------
// Kernel 1: fused  [build h_in] -> L1 GEMM -> LN -> ReLU -> L2 GEMM -> LN ->
//           ReLU -> masked pool (atomics into ws)
// grid = (B*J)/128 blocks of 256 threads; each block handles 128 rows of one b
// ---------------------------------------------------------------------------
__global__ __launch_bounds__(THREADS)
void token_mlp_pool_kernel(const float* __restrict__ x,
                           const int*   __restrict__ mask,
                           const int*   __restrict__ atse_index,
                           const float* __restrict__ Femb,    // [J, 32]
                           const float* __restrict__ Aemb,    // [A, 16]
                           const float* __restrict__ hW1,     // [49, 128]
                           const float* __restrict__ hb1,
                           const float* __restrict__ hg1,
                           const float* __restrict__ hB1,
                           const float* __restrict__ hW2,     // [128, 32]
                           const float* __restrict__ hb2,
                           const float* __restrict__ hg2,
                           const float* __restrict__ hB2,
                           float* __restrict__ pooled,        // [32, 32]
                           float* __restrict__ cnt)           // [32]
{
    __shared__ __align__(16) unsigned char smem[SMEM_BYTES];
    __bf16* sA  = (__bf16*)(smem + OFF_SA);   // [128][72]  h_in (bf16)
    __bf16* sW1 = (__bf16*)(smem + OFF_SW1);  // [128][72]  W1^T (bf16) [n][k]
    __bf16* sH1 = (__bf16*)(smem + OFF_SH1);  // [128][136] h1 (bf16)  (alias!)
    __bf16* sW2 = (__bf16*)(smem + OFF_SW2);  // [32][136]  W2^T (bf16)
    float*  sF  = (float*)(smem + OFF_F32);
    float* sHb1 = sF;            // 128
    float* sHg1 = sF + 128;      // 128
    float* sHB1 = sF + 256;      // 128
    float* sHb2 = sF + 384;      // 32
    float* sHg2 = sF + 416;      // 32
    float* sHB2 = sF + 448;      // 32
    float* sMask= sF + 480;      // 128
    float* sWsum= sF + 608;      // 8 waves * 32 cols

    const int tid  = threadIdx.x;
    const int lane = tid & 31;
    const int wave = tid >> 5;
    const int b    = blockIdx.x >> 7;            // 128 blocks per batch row
    const int j0   = (blockIdx.x & 127) * TILE_ROWS;

    // ---- Stage 0: build LDS tiles (bf16 conversions) ----------------------
    // h_in tile: [row][k]  k: 0 -> x, 1..32 -> Femb, 33..48 -> Aemb, pad 0
    for (int idx = tid; idx < TILE_ROWS * 64; idx += THREADS) {
        int row = idx >> 6, k = idx & 63;
        int j = j0 + row;
        float v = 0.0f;
        if (k == 0)       v = x[b * J_ + j];
        else if (k < 33)  v = Femb[j * D_ + (k - 1)];
        else if (k < 49)  v = Aemb[atse_index[j] * AE_ + (k - 33)];
        sA[row * STRIDE_A + k] = (__bf16)v;
    }
    // W1^T: [n][k]
    for (int idx = tid; idx < H_ * 64; idx += THREADS) {
        int n = idx >> 6, k = idx & 63;
        float v = (k < INDIM_) ? hW1[k * H_ + n] : 0.0f;
        sW1[n * STRIDE_A + k] = (__bf16)v;
    }
    // W2^T: [n][k]
    for (int idx = tid; idx < D_ * H_; idx += THREADS) {
        int n = idx >> 7, k = idx & 127;
        sW2[n * STRIDE_H + k] = (__bf16)hW2[k * D_ + n];
    }
    if (tid < 128) {
        sHb1[tid] = hb1[tid]; sHg1[tid] = hg1[tid]; sHB1[tid] = hB1[tid];
        sMask[tid] = (float)mask[b * J_ + j0 + tid];
    }
    if (tid < 32) { sHb2[tid] = hb2[tid]; sHg2[tid] = hg2[tid]; sHB2[tid] = hB2[tid]; }
    __syncthreads();

    // mask count for this tile (wave 0 only; sMask is read-only from here on)
    if (tid < 32) {
        float c = sMask[tid] + sMask[tid + 32] + sMask[tid + 64] + sMask[tid + 96];
        c += __shfl_xor(c, 1, 32);  c += __shfl_xor(c, 2, 32);
        c += __shfl_xor(c, 4, 32);  c += __shfl_xor(c, 8, 32);
        c += __shfl_xor(c, 16, 32);
        if (tid == 0) atomicAdd(&cnt[b], c);
    }

    // ---- Stage 1: layer1 GEMM  [16 x 64] @ [64 x 128] per wave ------------
    const int wbase = wave * 16;                 // this wave's row stripe
    const int lcol  = lane & 15;                 // N within a tile / M sub-row
    const int offA  = (lane >= 16) ? 8 : 0;      // A-frag K offset (ISA layout)
    const int offB  = (lane >= 16) ? 8 : 0;      // B-frag K offset (ISA layout)

    v8f acc[8] = {};
    {
        const __bf16* arow = sA + (wbase + lcol) * STRIDE_A;
        v16bf a0 = load_frag(arow + 0  + offA, 16);   // K = 0..31
        v16bf a1 = load_frag(arow + 32 + offA, 16);   // K = 32..63
#pragma unroll
        for (int t = 0; t < 8; ++t) {
            const __bf16* bcol = sW1 + (t * 16 + lcol) * STRIDE_A;
            acc[t] = wmma_bf16(a0, load_frag(bcol + 0  + offB, 16), acc[t]);
            acc[t] = wmma_bf16(a1, load_frag(bcol + 32 + offB, 16), acc[t]);
        }
    }
    __syncthreads();   // all waves done reading sA/sW1 before sH1 overwrites

    // ---- bias + LayerNorm(128) + ReLU -> sH1 (bf16) -----------------------
    {
#pragma unroll
        for (int t = 0; t < 8; ++t) {
            float bias = sHb1[t * 16 + lcol];
#pragma unroll
            for (int r = 0; r < 8; ++r) acc[t][r] += bias;
        }
        float mean8[8], rstd8[8];
#pragma unroll
        for (int r = 0; r < 8; ++r) {
            float s = 0.f, q = 0.f;
#pragma unroll
            for (int t = 0; t < 8; ++t) { float v = acc[t][r]; s += v; q += v * v; }
            // reduce across the 16 lanes holding this row's columns
            s += __shfl_xor(s, 1, 32); q += __shfl_xor(q, 1, 32);
            s += __shfl_xor(s, 2, 32); q += __shfl_xor(q, 2, 32);
            s += __shfl_xor(s, 4, 32); q += __shfl_xor(q, 4, 32);
            s += __shfl_xor(s, 8, 32); q += __shfl_xor(q, 8, 32);
            float mu = s * (1.0f / 128.0f);
            float var = q * (1.0f / 128.0f) - mu * mu;
            mean8[r] = mu;
            rstd8[r] = rsqrtf(var + EPS_);
        }
        const int rowAdd = (lane >= 16) ? 8 : 0;
#pragma unroll
        for (int t = 0; t < 8; ++t) {
            int col = t * 16 + lcol;
            float g = sHg1[col], bb = sHB1[col];
#pragma unroll
            for (int r = 0; r < 8; ++r) {
                float y = (acc[t][r] - mean8[r]) * rstd8[r] * g + bb;
                y = fmaxf(y, 0.0f);
                sH1[(wbase + r + rowAdd) * STRIDE_H + col] = (__bf16)y;
            }
        }
    }
    __syncthreads();

    // ---- Stage 2: layer2 GEMM  [16 x 128] @ [128 x 32] per wave -----------
    v8f acc2[2] = {};
    {
        const __bf16* arow = sH1 + (wbase + lcol) * STRIDE_H;
#pragma unroll
        for (int ks = 0; ks < 4; ++ks) {
            v16bf a = load_frag(arow + ks * 32 + offA, 16);
#pragma unroll
            for (int t = 0; t < 2; ++t) {
                const __bf16* bcol = sW2 + (t * 16 + lcol) * STRIDE_H;
                acc2[t] = wmma_bf16(a, load_frag(bcol + ks * 32 + offB, 16), acc2[t]);
            }
        }
    }

    // ---- bias + LayerNorm(32) + ReLU + masked pool ------------------------
    {
#pragma unroll
        for (int t = 0; t < 2; ++t) {
            float bias = sHb2[t * 16 + lcol];
#pragma unroll
            for (int r = 0; r < 8; ++r) acc2[t][r] += bias;
        }
        float mean8[8], rstd8[8];
#pragma unroll
        for (int r = 0; r < 8; ++r) {
            float v0 = acc2[0][r], v1 = acc2[1][r];
            float s = v0 + v1, q = v0 * v0 + v1 * v1;
            s += __shfl_xor(s, 1, 32); q += __shfl_xor(q, 1, 32);
            s += __shfl_xor(s, 2, 32); q += __shfl_xor(q, 2, 32);
            s += __shfl_xor(s, 4, 32); q += __shfl_xor(q, 4, 32);
            s += __shfl_xor(s, 8, 32); q += __shfl_xor(q, 8, 32);
            float mu = s * (1.0f / 32.0f);
            float var = q * (1.0f / 32.0f) - mu * mu;
            mean8[r] = mu;
            rstd8[r] = rsqrtf(var + EPS_);
        }
        const int rowAdd = (lane >= 16) ? 8 : 0;
        float p0 = 0.f, p1 = 0.f;
        float g0 = sHg2[lcol],      b0 = sHB2[lcol];
        float g1 = sHg2[16 + lcol], b1 = sHB2[16 + lcol];
#pragma unroll
        for (int r = 0; r < 8; ++r) {
            float m = sMask[wbase + r + rowAdd];
            float y0 = fmaxf((acc2[0][r] - mean8[r]) * rstd8[r] * g0 + b0, 0.0f);
            float y1 = fmaxf((acc2[1][r] - mean8[r]) * rstd8[r] * g1 + b1, 0.0f);
            p0 += m * y0;
            p1 += m * y1;
        }
        // combine the two half-waves (same column, other 8 rows)
        p0 += __shfl_xor(p0, 16, 32);
        p1 += __shfl_xor(p1, 16, 32);
        if (lane < 16) {
            sWsum[wave * 32 + lane]      = p0;
            sWsum[wave * 32 + 16 + lane] = p1;
        }
    }
    __syncthreads();
    if (tid < 32) {
        float s = 0.f;
#pragma unroll
        for (int w = 0; w < 8; ++w) s += sWsum[w * 32 + tid];
        atomicAdd(&pooled[b * D_ + tid], s);
    }
}

// ---------------------------------------------------------------------------
// Kernel 2: mean divide + encoder MLP (32 rows, tiny) -> mu / logvar
// single block of 256 threads, plain fp32
// ---------------------------------------------------------------------------
__global__ __launch_bounds__(THREADS)
void encoder_kernel(const float* __restrict__ pooled,   // [32, 32]
                    const float* __restrict__ cnt,      // [32]
                    const float* __restrict__ eW1,      // [32, 256]
                    const float* __restrict__ eb1,
                    const float* __restrict__ eg1,
                    const float* __restrict__ eB1,
                    const float* __restrict__ eW2,      // [256, 64]
                    const float* __restrict__ eb2,
                    const float* __restrict__ eg2,
                    const float* __restrict__ eB2,
                    float* __restrict__ out)            // [1024 mu | 1024 logvar]
{
    __shared__ float sC[B_ * D_];        // 32x32 pooled mean
    __shared__ float sZ[B_ * EH_];       // 32x256 stage-1 activations
    __shared__ float sZ2[B_ * 2 * LAT_]; // 32x64 stage-2 pre-LN
    __shared__ float sMu[B_], sRs[B_], sMu2[B_], sRs2[B_];

    const int tid = threadIdx.x;

    for (int i = tid; i < B_ * D_; i += THREADS) {
        float c = cnt[i >> 5];
        sC[i] = pooled[i] / fmaxf(c, 1.0f);
    }
    __syncthreads();

    // ---- e = c @ eW1 + eb1 ; each thread owns output column n = tid -------
    {
        const int n = tid;                       // 0..255
        float w[32];
#pragma unroll
        for (int k = 0; k < 32; ++k) w[k] = eW1[k * EH_ + n];
        float bb = eb1[n];
        for (int r = 0; r < B_; ++r) {
            float a = bb;
#pragma unroll
            for (int k = 0; k < 32; ++k) a += sC[r * 32 + k] * w[k];
            sZ[r * EH_ + n] = a;
        }
    }
    __syncthreads();
    if (tid < B_) {                              // LN stats over 256
        float s = 0.f, q = 0.f;
        for (int k = 0; k < EH_; ++k) { float v = sZ[tid * EH_ + k]; s += v; q += v * v; }
        float mu = s * (1.0f / EH_);
        sMu[tid] = mu;
        sRs[tid] = rsqrtf(q * (1.0f / EH_) - mu * mu + EPS_);
    }
    __syncthreads();
    {
        const int n = tid;
        float g = eg1[n], bb = eB1[n];
        for (int r = 0; r < B_; ++r) {
            float v = (sZ[r * EH_ + n] - sMu[r]) * sRs[r] * g + bb;
            sZ[r * EH_ + n] = fmaxf(v, 0.0f);
        }
    }
    __syncthreads();

    // ---- ml = e @ eW2 + eb2 ; 2048 outputs, 8 per thread -------------------
    for (int o = tid; o < B_ * 2 * LAT_; o += THREADS) {
        int r = o >> 6, n = o & 63;
        float a = eb2[n];
        for (int k = 0; k < EH_; ++k) a += sZ[r * EH_ + k] * eW2[k * (2 * LAT_) + n];
        sZ2[o] = a;
    }
    __syncthreads();
    if (tid < B_) {                              // LN stats over 64
        float s = 0.f, q = 0.f;
        for (int k = 0; k < 2 * LAT_; ++k) { float v = sZ2[tid * 64 + k]; s += v; q += v * v; }
        float mu = s * (1.0f / 64.0f);
        sMu2[tid] = mu;
        sRs2[tid] = rsqrtf(q * (1.0f / 64.0f) - mu * mu + EPS_);
    }
    __syncthreads();
    for (int o = tid; o < B_ * 2 * LAT_; o += THREADS) {
        int r = o >> 6, n = o & 63;
        float v = (sZ2[o] - sMu2[r]) * sRs2[r] * eg2[n] + eB2[n];
        v = fmaxf(v, 0.0f);
        if (n < LAT_) out[r * LAT_ + n]           = v;   // mu
        else          out[B_ * LAT_ + r * LAT_ + (n - LAT_)] = v;   // logvar
    }
}

// ---------------------------------------------------------------------------
extern "C" void kernel_launch(void* const* d_in, const int* in_sizes, int n_in,
                              void* d_out, int out_size, void* d_ws, size_t ws_size,
                              hipStream_t stream) {
    const float* x    = (const float*)d_in[0];
    const int*   mask = (const int*)  d_in[1];
    const int*   aidx = (const int*)  d_in[2];
    const float* Femb = (const float*)d_in[3];
    const float* Aemb = (const float*)d_in[4];
    const float* hW1  = (const float*)d_in[5];
    const float* hb1  = (const float*)d_in[6];
    const float* hg1  = (const float*)d_in[7];
    const float* hB1  = (const float*)d_in[8];
    const float* hW2  = (const float*)d_in[9];
    const float* hb2  = (const float*)d_in[10];
    const float* hg2  = (const float*)d_in[11];
    const float* hB2  = (const float*)d_in[12];
    const float* eW1  = (const float*)d_in[13];
    const float* eb1  = (const float*)d_in[14];
    const float* eg1  = (const float*)d_in[15];
    const float* eB1  = (const float*)d_in[16];
    const float* eW2  = (const float*)d_in[17];
    const float* eb2  = (const float*)d_in[18];
    const float* eg2  = (const float*)d_in[19];
    const float* eB2  = (const float*)d_in[20];

    float* ws     = (float*)d_ws;
    float* pooled = ws;            // [32*32]
    float* cnt    = ws + B_ * D_;  // [32]
    float* out    = (float*)d_out;

    const int nzero = B_ * D_ + B_;
    zero_ws_kernel<<<(nzero + 255) / 256, 256, 0, stream>>>(ws, nzero);

    const int nblocks = (B_ * J_) / TILE_ROWS;   // 4096
    token_mlp_pool_kernel<<<nblocks, THREADS, 0, stream>>>(
        x, mask, aidx, Femb, Aemb,
        hW1, hb1, hg1, hB1, hW2, hb2, hg2, hB2,
        pooled, cnt);

    encoder_kernel<<<1, THREADS, 0, stream>>>(
        pooled, cnt, eW1, eb1, eg1, eB1, eW2, eb2, eg2, eB2, out);
}
